// LoRAPlusCompressor_65077344469080
// MI455X (gfx1250) — matrix operand order repaired and checked by reading, not compiled, not run on hardware
//
#include <hip/hip_runtime.h>
#include <hip/hip_bf16.h>
#include <math.h>

// ---------------- problem constants (match reference) ----------------
#define Hdim 2048
#define HHdim 1024          // H/2
#define Bsz  4
#define Sdim 4096
#define ROWS (Bsz * Sdim)   // 16384 rows per tensor
#define TILE_M 16
#define LDS_STRIDE (Hdim + 4)   // padded fp32 row stride in LDS (bank-conflict avoidance)

// ws layout (in floats)
#define WS_PART 0                          // 16 * (B*H) = 131072 floats
#define WS_HPRE (16 * Bsz * Hdim)          // 16 * B * HH = 65536 floats
#define WS_RANK (WS_HPRE + 16 * Bsz * HHdim)  // 1 int

typedef __attribute__((ext_vector_type(16))) __bf16 v16bf;
typedef __attribute__((ext_vector_type(8)))  float  v8f;

// ---------------- kernel 1: partial column sums of keys over S ----------------
// grid (16 s-chunks, 32 col-blocks), 256 threads. Deterministic (fixed-order).
__global__ void colsum_partial(const float* __restrict__ keys, float* __restrict__ ws) {
    int sc  = blockIdx.x;                       // s-chunk: 256 rows each
    int c   = blockIdx.y * 256 + threadIdx.x;   // 0 .. B*H-1
    int b   = c >> 11;                          // / 2048
    int h   = c & (Hdim - 1);
    const float* p = keys + ((size_t)b * Sdim + (size_t)sc * 256) * Hdim + h;
    float acc = 0.f;
    for (int i = 0; i < 256; ++i) acc += p[(size_t)i * Hdim];
    ws[WS_PART + sc * (Bsz * Hdim) + c] = acc;
}

// ---------------- kernel 2a: layer-1 MLP partials ----------------
// grid (16 k-chunks, 8 j-blocks), 128 threads.
__global__ void mlp1_partial(const float* __restrict__ w1, float* __restrict__ ws) {
    __shared__ float xm[Bsz][128];
    int kc  = blockIdx.x;       // k chunk of 128
    int jb  = blockIdx.y;       // j block of 128
    int tid = threadIdx.x;

    // build x_mean for this k chunk (sum of 16 partials / S)
    for (int v = tid; v < Bsz * 128; v += 128) {
        int b  = v >> 7;
        int kk = v & 127;
        int k  = kc * 128 + kk;
        float s = 0.f;
        for (int p = 0; p < 16; ++p)
            s += ws[WS_PART + p * (Bsz * Hdim) + b * Hdim + k];
        xm[b][kk] = s * (1.0f / Sdim);
    }
    __syncthreads();

    int j = jb * 128 + tid;
    float a0 = 0.f, a1 = 0.f, a2 = 0.f, a3 = 0.f;
    for (int kk = 0; kk < 128; ++kk) {
        float w = w1[(size_t)(kc * 128 + kk) * HHdim + j];
        a0 = fmaf(xm[0][kk], w, a0);
        a1 = fmaf(xm[1][kk], w, a1);
        a2 = fmaf(xm[2][kk], w, a2);
        a3 = fmaf(xm[3][kk], w, a3);
    }
    float* hp = ws + WS_HPRE + kc * (Bsz * HHdim);
    hp[0 * HHdim + j] = a0;
    hp[1 * HHdim + j] = a1;
    hp[2 * HHdim + j] = a2;
    hp[3 * HHdim + j] = a3;
}

// ---------------- kernel 2b: finish MLP, sigmoid, threshold -> rank_idx ----------------
__global__ void mlp2_rank(const float* __restrict__ b1, const float* __restrict__ w2,
                          const float* __restrict__ b2, float* __restrict__ ws) {
    __shared__ float red[Bsz][1024];
    int j = threadIdx.x;           // 0..1023
    float bb1 = b1[j], ww2 = w2[j];
    for (int b = 0; b < Bsz; ++b) {
        float h = bb1;
        for (int kc = 0; kc < 16; ++kc)
            h += ws[WS_HPRE + kc * (Bsz * HHdim) + b * HHdim + j];
        h = fmaxf(h, 0.f);         // ReLU
        red[b][j] = h * ww2;
    }
    __syncthreads();
    for (int off = 512; off > 0; off >>= 1) {
        if (j < off)
            for (int b = 0; b < Bsz; ++b) red[b][j] += red[b][j + off];
        __syncthreads();
    }
    if (j == 0) {
        float avg = 0.f;
        for (int b = 0; b < Bsz; ++b)
            avg += 1.0f / (1.0f + expf(-(red[b][0] + b2[0])));
        avg *= 0.25f;
        int rank_idx = (avg >= 0.3f ? 1 : 0) + (avg >= 0.7f ? 1 : 0);
        ((int*)(ws + WS_RANK))[0] = rank_idx;
    }
}

// ---------------- kernel 3: out = X + (X @ A) @ B via bf16 WMMA ----------------
// grid 2048 (1024 tiles x 2 tensors), 128 threads = 4 wave32s.
__global__ __launch_bounds__(128) void lora_apply(
    const float* __restrict__ keys, const float* __restrict__ values,
    const float* __restrict__ kA0, const float* __restrict__ kB0,
    const float* __restrict__ vA0, const float* __restrict__ vB0,
    const float* __restrict__ kA1, const float* __restrict__ kB1,
    const float* __restrict__ vA1, const float* __restrict__ vB1,
    const float* __restrict__ kA2, const float* __restrict__ kB2,
    const float* __restrict__ vA2, const float* __restrict__ vB2,
    float* __restrict__ out, const float* __restrict__ ws) {

    __shared__ float xt[TILE_M * LDS_STRIDE];   // 16 x 2048 fp32 tile (padded) ~128 KB
    __shared__ float tpart[4][16][16];          // per-wave partial T

    int rank_idx = ((const int*)(ws + WS_RANK))[0];
    int r = 4 << rank_idx;                      // 4, 8, 16

    int t    = blockIdx.x >> 10;                // 0 = keys, 1 = values
    int tile = blockIdx.x & 1023;
    const float* X = t ? values : keys;
    const float* Amat;
    const float* Bmat;
    if (t == 0) {
        Amat = rank_idx == 0 ? kA0 : (rank_idx == 1 ? kA1 : kA2);
        Bmat = rank_idx == 0 ? kB0 : (rank_idx == 1 ? kB1 : kB2);
    } else {
        Amat = rank_idx == 0 ? vA0 : (rank_idx == 1 ? vA1 : vA2);
        Bmat = rank_idx == 0 ? vB0 : (rank_idx == 1 ? vB1 : vB2);
    }
    size_t row0 = (size_t)tile * TILE_M;

    int tid = threadIdx.x;
    // ---- async memory->LDS fill of the 16x2048 fp32 tile (ASYNCcnt path) ----
    // 8192 x b128 transfers; each of the 128 threads issues 64.
    {
        const char* gbase = (const char*)(X + row0 * Hdim);
        for (int f = tid; f < TILE_M * (Hdim / 4); f += 128) {
            int row = f >> 9;                       // / 512 float4 per row
            int c4  = f & 511;
            unsigned int       ldsoff = (unsigned int)(size_t)&xt[row * LDS_STRIDE + c4 * 4];
            unsigned long long gaddr  = (unsigned long long)(size_t)(gbase + (size_t)f * 16);
            asm volatile("global_load_async_to_lds_b128 %0, %1, off"
                         :: "v"(ldsoff), "v"(gaddr) : "memory");
        }
        asm volatile("s_wait_asynccnt 0\n\ts_wait_xcnt 0" ::: "memory");
    }
    __syncthreads();

    int  wave = tid >> 5;
    int  lane = tid & 31;
    int  hl   = lane & 15;
    bool hi   = lane >= 16;
    int  koffA = hi ? 8 : 0;                    // 16-bit A operand K sub-offset
    float colmask = (hl < r) ? 1.0f : 0.0f;
    int   ncl     = (hl < r) ? hl : 0;          // clamped column (keeps loads in-bounds, branchless)

    // ---- stage 1: partial T = X(16xK) @ A(Kxr), K split over 4 waves ----
    v8f acc = {};
    for (int it = 0; it < 16; ++it) {
        int kbase = wave * 512 + it * 32;
        // A operand: X rows (16-bit A 16x32 layout: lane<16 -> K 0-7,16-23; lane>=16 -> K 8-15,24-31)
        v16bf a;
        {
            const float* xr = &xt[hl * LDS_STRIDE + kbase + koffA];
            float4 g0 = *(const float4*)(xr + 0);
            float4 g1 = *(const float4*)(xr + 4);
            float4 g2 = *(const float4*)(xr + 16);
            float4 g3 = *(const float4*)(xr + 20);
            a[0]  = (__bf16)g0.x; a[1]  = (__bf16)g0.y; a[2]  = (__bf16)g0.z; a[3]  = (__bf16)g0.w;
            a[4]  = (__bf16)g1.x; a[5]  = (__bf16)g1.y; a[6]  = (__bf16)g1.z; a[7]  = (__bf16)g1.w;
            a[8]  = (__bf16)g2.x; a[9]  = (__bf16)g2.y; a[10] = (__bf16)g2.z; a[11] = (__bf16)g2.w;
            a[12] = (__bf16)g3.x; a[13] = (__bf16)g3.y; a[14] = (__bf16)g3.z; a[15] = (__bf16)g3.w;
        }
        // B operand: A_lora[k][n], n = hl; branchless zero-pad for n >= r
        v16bf bop;
        {
            int kb = kbase + (hi ? 8 : 0);
            const float* ap = Amat + (size_t)kb * r + ncl;
            #pragma unroll
            for (int e = 0; e < 16; ++e)
                bop[e] = (__bf16)(ap[(size_t)(e + (e & 8)) * r] * colmask);
        }
        acc = __builtin_amdgcn_wmma_f32_16x16x32_bf16(false, a, false, bop,
                                                      (short)0, acc, false, false);
    }
    // C/D layout: VGPR i -> M = i (lanes 0-15) or 8+i (lanes 16-31), N = hl
    #pragma unroll
    for (int i = 0; i < 8; ++i) {
        int M = hi ? (8 + i) : i;
        tpart[wave][M][hl] = acc[i];
    }
    __syncthreads();

    // ---- stage 2: out = X + T(16x16) @ B(16xH), K padded 16->32 with zeros ----
    v16bf aT;
    #pragma unroll
    for (int e = 0; e < 8; ++e) {
        int K = koffA + e;
        float s = tpart[0][hl][K] + tpart[1][hl][K] + tpart[2][hl][K] + tpart[3][hl][K];
        aT[e]     = (__bf16)s;       // K in [0,16)
        aT[e + 8] = (__bf16)0.0f;    // K in [16,32) padding
    }

    int kb2 = hi ? 8 : 0;
    float* op = out + (size_t)t * ROWS * Hdim + row0 * Hdim;
    for (int nt = wave; nt < Hdim / 16; nt += 4) {
        int n0 = nt * 16;
        // B operand: B_lora[k][n0+hl]; branchless zero for k >= r (aT is zero there too)
        v16bf bop;
        #pragma unroll
        for (int e = 0; e < 16; ++e) {
            int   k  = kb2 + e + (e & 8);
            int   kcl = (k < r) ? k : 0;
            float km  = (k < r) ? 1.0f : 0.0f;
            bop[e] = (__bf16)(Bmat[(size_t)kcl * Hdim + n0 + hl] * km);
        }
        // preload C with the fp32 X tile -> residual add is free
        v8f c;
        #pragma unroll
        for (int i = 0; i < 8; ++i) {
            int M = hi ? (8 + i) : i;
            c[i] = xt[M * LDS_STRIDE + n0 + hl];
        }
        c = __builtin_amdgcn_wmma_f32_16x16x32_bf16(false, aT, false, bop,
                                                    (short)0, c, false, false);
        #pragma unroll
        for (int i = 0; i < 8; ++i) {
            int M = hi ? (8 + i) : i;
            op[(size_t)M * Hdim + n0 + hl] = c[i];
        }
    }
}

// ---------------- launch ----------------
extern "C" void kernel_launch(void* const* d_in, const int* in_sizes, int n_in,
                              void* d_out, int out_size, void* d_ws, size_t ws_size,
                              hipStream_t stream) {
    const float* keys   = (const float*)d_in[0];
    const float* values = (const float*)d_in[1];
    const float* w1     = (const float*)d_in[2];
    const float* b1     = (const float*)d_in[3];
    const float* w2     = (const float*)d_in[4];
    const float* b2     = (const float*)d_in[5];
    const float* kA0 = (const float*)d_in[6],  *kB0 = (const float*)d_in[7];
    const float* vA0 = (const float*)d_in[8],  *vB0 = (const float*)d_in[9];
    const float* kA1 = (const float*)d_in[10], *kB1 = (const float*)d_in[11];
    const float* vA1 = (const float*)d_in[12], *vB1 = (const float*)d_in[13];
    const float* kA2 = (const float*)d_in[14], *kB2 = (const float*)d_in[15];
    const float* vA2 = (const float*)d_in[16], *vB2 = (const float*)d_in[17];
    float* out = (float*)d_out;
    float* ws  = (float*)d_ws;

    colsum_partial<<<dim3(16, 32), 256, 0, stream>>>(keys, ws);
    mlp1_partial<<<dim3(16, 8), 128, 0, stream>>>(w1, ws);
    mlp2_rank<<<1, 1024, 0, stream>>>(b1, w2, b2, ws);
    lora_apply<<<2048, 128, 0, stream>>>(keys, values,
                                         kA0, kB0, vA0, vB0,
                                         kA1, kB1, vA1, vB1,
                                         kA2, kB2, vA2, vB2,
                                         out, ws);
}
